// NonLocalBlock3D_53094385713235
// MI455X (gfx1250) — compile-verified, matrix operand order checked by reading, not run on hardware
//
#include <hip/hip_runtime.h>
#include <hip/hip_bf16.h>
#include <stdint.h>

// Problem constants (reference: B, C, HID, T, H, W = 2, 256, 128, 8, 32, 32)
#define B_   2
#define C_   256
#define HID_ 128
#define N_   8192   // T*H*W

typedef __attribute__((ext_vector_type(16))) __bf16 v16bf;
typedef __attribute__((ext_vector_type(2)))  __bf16 bf16x2;
typedef __attribute__((ext_vector_type(8)))  float  v8f;
typedef __attribute__((ext_vector_type(4)))  unsigned int u32x4;
typedef __attribute__((ext_vector_type(2)))  unsigned int u32x2;
typedef __attribute__((ext_vector_type(4)))  float  f32x4;
typedef __attribute__((ext_vector_type(2)))  float  f32x2;

union Frag {
    v16bf v;
    unsigned int u[8];
    u32x4 q[2];
};

// Packed f32x2 -> bf16x2 (single v_cvt_pk_bf16_f32 via build-vector pattern).
static __device__ __forceinline__ unsigned int packbf(float a, float b) {
    bf16x2 t;
    t[0] = (__bf16)a;
    t[1] = (__bf16)b;
    return __builtin_bit_cast(unsigned int, t);
}
static __device__ __forceinline__ v8f zero8() {
    v8f z;
#pragma unroll
    for (int i = 0; i < 8; ++i) z[i] = 0.0f;
    return z;
}

// CDNA5 16-bit fragment K-pattern (ISA 7.12.2, wave32):
//   VGPR j holds packed K = (j/4)*16 + half*8 + (j%4)*2 and K+1.
// Per lane this is two contiguous 16-byte spans -> two b128 loads.
static __device__ __forceinline__ v16bf load_frag_bf(const __bf16* rowptr, int half) {
    Frag f;
    f.q[0] = *(const u32x4*)(rowptr + half * 8);        // K = half*8 + 0..7
    f.q[1] = *(const u32x4*)(rowptr + 16 + half * 8);   // K = 16 + half*8 + 0..7
    return f.v;
}

static __device__ __forceinline__ v8f wmma_bf(v16bf a, v16bf b, v8f c) {
    return __builtin_amdgcn_wmma_f32_16x16x32_bf16(false, a, false, b, (short)0, c, false, false);
}

// ---------------------------------------------------------------------------
// Kernel 0a: f32 -> bf16 weight conversion (pairs; n even).
// ---------------------------------------------------------------------------
__global__ void k_cvt(const float* __restrict__ src, __bf16* __restrict__ dst) {
    int i = blockIdx.x * blockDim.x + threadIdx.x;
    f32x2 v = *(const f32x2*)(src + 2 * i);
    *(unsigned int*)(dst + 2 * i) = packbf(v[0], v[1]);
}

// ---------------------------------------------------------------------------
// Kernel 0b: x[b][c][n] f32 -> xT[b][n][c] bf16, 32x32 tiles via LDS.
// ---------------------------------------------------------------------------
__global__ void k_xpose(const float* __restrict__ x, __bf16* __restrict__ xT) {
    __shared__ __bf16 tile[32 * 34];  // even pad: transposed reads stay dword-aligned
    const int blk = blockIdx.x;       // B*(C/32)*(N/256... ) = 2*8*256 = 4096
    const int nt = blk & 255;         // N/32
    const int ct = (blk >> 8) & 7;    // C/32
    const int b  = blk >> 11;
    const int tid = threadIdx.x;      // 256 threads
    const int c0 = tid >> 3;          // 0..31
    const int n0 = (tid & 7) << 2;    // 0,4,..,28
    f32x4 v = *(const f32x4*)(x + ((size_t)(b * C_ + ct * 32 + c0)) * N_ + nt * 32 + n0);
#pragma unroll
    for (int i = 0; i < 4; ++i) tile[(n0 + i) * 34 + c0] = (__bf16)v[i];
    __syncthreads();
    const int n1 = tid >> 3;          // 0..31
    const int c1 = (tid & 7) << 2;    // 0,4,..,28
    u32x2 w;
    w[0] = *(const unsigned int*)&tile[n1 * 34 + c1];
    w[1] = *(const unsigned int*)&tile[n1 * 34 + c1 + 2];
    *(u32x2*)(xT + ((size_t)(b * N_ + nt * 32 + n1)) * C_ + ct * 32 + c1) = w;
}

// ---------------------------------------------------------------------------
// Kernel 1: thetaT/phiT projections.  D(16n x 16o) = A(xT 16n x 32c) * B(W^T 32c x 16o)
// Output layout [B][N][HID] bf16 (K=HID contiguous for later fragments).
// ---------------------------------------------------------------------------
__global__ void k_proj_qk(const __bf16* __restrict__ xT, const __bf16* __restrict__ twb,
                          const __bf16* __restrict__ pwb,
                          __bf16* __restrict__ thetaT, __bf16* __restrict__ phiT) {
    const int lane = threadIdx.x & 31;
    const int idx = lane & 15, half = lane >> 4;
    const int wid = (blockIdx.x * blockDim.x + threadIdx.x) >> 5;  // 0..8191
    const int ot = wid & 7;
    const int nt = (wid >> 3) & 511;
    const int b  = wid >> 12;

    const __bf16* xrow = xT + ((size_t)(b * N_ + nt * 16 + idx)) * C_;
    const __bf16* trow = twb + (size_t)(ot * 16 + idx) * C_;
    const __bf16* prow = pwb + (size_t)(ot * 16 + idx) * C_;
    v8f accT = zero8(), accP = zero8();
#pragma unroll
    for (int cc = 0; cc < C_; cc += 32) {
        v16bf a = load_frag_bf(xrow + cc, half);
        accT = wmma_bf(a, load_frag_bf(trow + cc, half), accT);
        accP = wmma_bf(a, load_frag_bf(prow + cc, half), accP);
    }
#pragma unroll
    for (int v = 0; v < 8; ++v) {
        int n = nt * 16 + v + 8 * half;                              // D row (M)
        size_t off = ((size_t)(b * N_ + n)) * HID_ + ot * 16 + idx;  // D col = idx
        thetaT[off] = (__bf16)accT[v];
        phiT[off]   = (__bf16)accP[v];
    }
}

// ---------------------------------------------------------------------------
// Kernel 2: g projection.  D(16o x 16n) = A(g_w 16o x 32c) * B(x 32c x 16n)
// Output layout [B][HID][N] f32 (scaled by 1/Z later).
// ---------------------------------------------------------------------------
__global__ void k_proj_g(const __bf16* __restrict__ xT, const __bf16* __restrict__ gwb,
                         float* __restrict__ gf) {
    const int lane = threadIdx.x & 31;
    const int idx = lane & 15, half = lane >> 4;
    const int wid = (blockIdx.x * blockDim.x + threadIdx.x) >> 5;
    const int ot = wid & 7;
    const int nt = (wid >> 3) & 511;
    const int b  = wid >> 12;

    const __bf16* grow = gwb + (size_t)(ot * 16 + idx) * C_;
    const __bf16* xrow = xT + ((size_t)(b * N_ + nt * 16 + idx)) * C_;  // B: lane = n col
    v8f acc = zero8();
#pragma unroll
    for (int cc = 0; cc < C_; cc += 32)
        acc = wmma_bf(load_frag_bf(grow + cc, half), load_frag_bf(xrow + cc, half), acc);
#pragma unroll
    for (int v = 0; v < 8; ++v) {
        int o = ot * 16 + v + 8 * half;
        gf[((size_t)(b * HID_ + o)) * N_ + nt * 16 + idx] = acc[v];
    }
}

// ---------------------------------------------------------------------------
// Kernel 3: pass A — per-key column softmax stats over the query axis.
// One wave owns a 16-key tile; streams all 512 q-tiles of S(16q x 16k) =
// A(thetaT) * B(phiT), online max/sum with a shfl_xor(16) half-fold.
// ---------------------------------------------------------------------------
__global__ void k_stats(const __bf16* __restrict__ thetaT, const __bf16* __restrict__ phiT,
                        float* __restrict__ mArr, float* __restrict__ zArr) {
    const int lane = threadIdx.x & 31;
    const int idx = lane & 15, half = lane >> 4;
    const int wid = (blockIdx.x * blockDim.x + threadIdx.x) >> 5;  // 0..1023
    const int kt = wid & 511;
    const int b  = wid >> 9;

    v16bf bf[4];
    const __bf16* krow = phiT + ((size_t)(b * N_ + kt * 16 + idx)) * HID_;
#pragma unroll
    for (int c = 0; c < 4; ++c) bf[c] = load_frag_bf(krow + c * 32, half);

    float m = -1e30f, Z = 0.0f;
    for (int qt = 0; qt < N_ / 16; ++qt) {
        const __bf16* arow = thetaT + ((size_t)(b * N_ + qt * 16 + idx)) * HID_;
        v8f acc = zero8();
#pragma unroll
        for (int c = 0; c < 4; ++c) acc = wmma_bf(load_frag_bf(arow + c * 32, half), bf[c], acc);

        float tm = acc[0];
#pragma unroll
        for (int v = 1; v < 8; ++v) tm = fmaxf(tm, acc[v]);
        tm = fmaxf(tm, __shfl_xor(tm, 16, 32));  // other 8 q-rows (same k col)
        float mn = fmaxf(m, tm);
        float s = 0.0f;
#pragma unroll
        for (int v = 0; v < 8; ++v) s += __expf(acc[v] - mn);
        s += __shfl_xor(s, 16, 32);
        Z = Z * __expf(m - mn) + s;
        m = mn;
    }
    if (lane < 16) {
        mArr[b * N_ + kt * 16 + lane] = m;
        zArr[b * N_ + kt * 16 + lane] = Z;
    }
}

// ---------------------------------------------------------------------------
// Kernel 4: g' = g / Z_k, cast to bf16 (layout [B][HID][N]).
// ---------------------------------------------------------------------------
__global__ void k_scale_g(const float* __restrict__ gf, const float* __restrict__ zArr,
                          __bf16* __restrict__ gp) {
    int i = blockIdx.x * blockDim.x + threadIdx.x;  // over (B*HID*N)/2 pairs
    int i2 = 2 * i;
    int k = i2 & (N_ - 1);
    int b = i2 >> 20;  // HID_*N_ = 2^20
    f32x2 g = *(const f32x2*)(gf + i2);
    f32x2 z = *(const f32x2*)(zArr + b * N_ + k);   // k even, pairs share row
    *(unsigned int*)(gp + i2) = packbf(g[0] / z[0], g[1] / z[1]);
}

// ---------------------------------------------------------------------------
// Kernel 5: pass B — out^T(16h x 16q) += A(g'^T 16h x 32k) * B(P^T 32k x 16q).
// S^T tiles are computed so the WMMA D layout (lane = q col, k rows in VGPRs)
// maps onto the P^T B-fragment k-pattern with only an in-lane exp+bf16 pack.
// ---------------------------------------------------------------------------
__global__ void k_attn_out(const __bf16* __restrict__ thetaT, const __bf16* __restrict__ phiT,
                           const __bf16* __restrict__ gp, const float* __restrict__ mArr,
                           __bf16* __restrict__ outb) {
    const int lane = threadIdx.x & 31;
    const int idx = lane & 15, half = lane >> 4;
    const int wid = (blockIdx.x * blockDim.x + threadIdx.x) >> 5;  // 0..1023
    const int qt = wid & 511;
    const int b  = wid >> 9;

    v16bf bq[4];  // theta for this q-tile, reused for every key tile
    const __bf16* qrow = thetaT + ((size_t)(b * N_ + qt * 16 + idx)) * HID_;
#pragma unroll
    for (int c = 0; c < 4; ++c) bq[c] = load_frag_bf(qrow + c * 32, half);

    v8f acc[8];
#pragma unroll
    for (int h = 0; h < 8; ++h) acc[h] = zero8();

    const float* mb = mArr + b * N_;
    for (int kb = 0; kb < N_; kb += 32) {
        Frag pfrag;  // P^T B operand: 32 keys x 16 queries
#pragma unroll
        for (int s = 0; s < 2; ++s) {
            const int kt = kb + 16 * s;
            const __bf16* krow = phiT + ((size_t)(b * N_ + kt + idx)) * HID_;
            v8f st = zero8();  // S^T tile: rows k = v + 8*half, col q = idx
#pragma unroll
            for (int c = 0; c < 4; ++c)
                st = wmma_bf(load_frag_bf(krow + c * 32, half), bq[c], st);
#pragma unroll
            for (int j = 0; j < 4; ++j) {
                f32x2 mm = *(const f32x2*)(mb + kt + 2 * j + 8 * half);
                float p0 = __expf(st[2 * j]     - mm[0]);
                float p1 = __expf(st[2 * j + 1] - mm[1]);
                pfrag.u[4 * s + j] = packbf(p0, p1);  // k = 16s + 8*half + 2j (+1)
            }
        }
#pragma unroll
        for (int ht = 0; ht < 8; ++ht) {
            const __bf16* grow = gp + ((size_t)(b * HID_ + ht * 16 + idx)) * N_ + kb;
            acc[ht] = wmma_bf(load_frag_bf(grow, half), pfrag.v, acc[ht]);
        }
    }
    // Store out as [B][N][HID] bf16: per lane, 8 consecutive h per h-tile -> one 16B store.
#pragma unroll
    for (int ht = 0; ht < 8; ++ht) {
        u32x4 w;
#pragma unroll
        for (int j = 0; j < 4; ++j) w[j] = packbf(acc[ht][2 * j], acc[ht][2 * j + 1]);
        size_t off = ((size_t)(b * N_ + qt * 16 + idx)) * HID_ + ht * 16 + 8 * half;
        *(u32x4*)(outb + off) = w;
    }
}

// ---------------------------------------------------------------------------
// Kernel 6: y = x + mask_w @ out^T.  D(16c x 16n) = A(mask_w 16c x 32h) * B(out 32h x 16n)
// ---------------------------------------------------------------------------
__global__ void k_final(const float* __restrict__ x, const __bf16* __restrict__ mwb,
                        const __bf16* __restrict__ outb, float* __restrict__ y) {
    const int lane = threadIdx.x & 31;
    const int idx = lane & 15, half = lane >> 4;
    const int wid = (blockIdx.x * blockDim.x + threadIdx.x) >> 5;  // 0..16383
    const int ct = wid & 15;
    const int nt = (wid >> 4) & 511;
    const int b  = wid >> 13;

    const __bf16* arow = mwb + (size_t)(ct * 16 + idx) * HID_;
    const __bf16* brow = outb + ((size_t)(b * N_ + nt * 16 + idx)) * HID_;
    v8f acc = zero8();
#pragma unroll
    for (int hc = 0; hc < HID_; hc += 32)
        acc = wmma_bf(load_frag_bf(arow + hc, half), load_frag_bf(brow + hc, half), acc);
#pragma unroll
    for (int v = 0; v < 8; ++v) {
        int c = ct * 16 + v + 8 * half;
        size_t off = ((size_t)(b * C_ + c)) * N_ + nt * 16 + idx;
        y[off] = x[off] + acc[v];
    }
}

// ---------------------------------------------------------------------------
extern "C" void kernel_launch(void* const* d_in, const int* in_sizes, int n_in,
                              void* d_out, int out_size, void* d_ws, size_t ws_size,
                              hipStream_t stream) {
    (void)in_sizes; (void)n_in; (void)out_size; (void)ws_size;
    const float* x  = (const float*)d_in[0];
    const float* tw = (const float*)d_in[1];
    const float* pw = (const float*)d_in[2];
    const float* gw = (const float*)d_in[3];
    const float* mw = (const float*)d_in[4];
    float* y = (float*)d_out;

    char* ws = (char*)d_ws;
    const size_t szNC_bf = (size_t)B_ * N_ * C_ * sizeof(__bf16);     // 8 MB
    const size_t szNH_bf = (size_t)B_ * N_ * HID_ * sizeof(__bf16);   // 4 MB
    const size_t szHN_f  = (size_t)B_ * HID_ * N_ * sizeof(float);    // 8 MB
    const size_t szK_f   = (size_t)B_ * N_ * sizeof(float);           // 64 KB
    const size_t szW_bf  = (size_t)HID_ * C_ * sizeof(__bf16);        // 64 KB
    size_t o = 0;
    __bf16* xT     = (__bf16*)(ws + o); o += szNC_bf;
    __bf16* thetaT = (__bf16*)(ws + o); o += szNH_bf;
    __bf16* phiT   = (__bf16*)(ws + o); o += szNH_bf;
    float*  gf     = (float*)(ws + o);  o += szHN_f;
    __bf16* gp     = (__bf16*)(ws + o); o += szNH_bf;
    __bf16* outb   = (__bf16*)(ws + o); o += szNH_bf;
    float*  mArr   = (float*)(ws + o);  o += szK_f;
    float*  zArr   = (float*)(ws + o);  o += szK_f;
    __bf16* twb    = (__bf16*)(ws + o); o += szW_bf;
    __bf16* pwb    = (__bf16*)(ws + o); o += szW_bf;
    __bf16* gwb    = (__bf16*)(ws + o); o += szW_bf;
    __bf16* mwb    = (__bf16*)(ws + o); o += szW_bf;

    const int TPB = 128;  // 4 waves per block
    // 0) one-time bf16 conversions (weights: HID*C = 32768 elems each)
    k_cvt<<<(HID_ * C_ / 2) / 256, 256, 0, stream>>>(tw, twb);
    k_cvt<<<(HID_ * C_ / 2) / 256, 256, 0, stream>>>(pw, pwb);
    k_cvt<<<(HID_ * C_ / 2) / 256, 256, 0, stream>>>(gw, gwb);
    k_cvt<<<(HID_ * C_ / 2) / 256, 256, 0, stream>>>(mw, mwb);
    k_xpose<<<B_ * (C_ / 32) * (N_ / 32), 256, 0, stream>>>(x, xT);
    // 1) projections: B*(N/16)*(HID/16) = 8192 wave-tiles
    k_proj_qk<<<8192 / (TPB / 32), TPB, 0, stream>>>(xT, twb, pwb, thetaT, phiT);
    k_proj_g <<<8192 / (TPB / 32), TPB, 0, stream>>>(xT, gwb, gf);
    // 2) column softmax stats: B*(N/16) = 1024 wave-tiles
    k_stats<<<1024 / (TPB / 32), TPB, 0, stream>>>(thetaT, phiT, mArr, zArr);
    // 3) g / Z (pairs)
    k_scale_g<<<(B_ * HID_ * N_ / 2) / 256, 256, 0, stream>>>(gf, zArr, gp);
    // 4) attention output: B*(N/16) = 1024 wave-tiles
    k_attn_out<<<1024 / (TPB / 32), TPB, 0, stream>>>(thetaT, phiT, gp, mArr, outb);
    // 5) residual + mask projection: B*(N/16)*(C/16) = 16384 wave-tiles
    k_final<<<16384 / (TPB / 32), TPB, 0, stream>>>(x, mwb, outb, y);
}
